// LiteMLA_58025008169523
// MI455X (gfx1250) — matrix operand order repaired and checked by reading, not compiled, not run on hardware
//
#include <hip/hip_runtime.h>
#include <math.h>

typedef __attribute__((ext_vector_type(16))) __bf16 v16bf;
typedef __attribute__((ext_vector_type(8)))  __bf16 v8bf;
typedef __attribute__((ext_vector_type(8)))  float  v8f;

#define NPIX 3136   // 56*56
#define HH   56
#define WW   56
#define EPSF 1e-15f

__device__ __forceinline__ __bf16 f2bf(float f) {
    union { float f; unsigned u; } x; x.f = f;
    unsigned r = x.u + 0x7FFFu + ((x.u >> 16) & 1u);   // round-to-nearest-even
    union { unsigned short s; __bf16 b; } y;
    y.s = (unsigned short)(r >> 16);
    return y.b;
}

// ---------------------------------------------------------------------------
// P0: elementwise fp32 -> bf16 (weights)
// ---------------------------------------------------------------------------
__global__ void __launch_bounds__(256) cvt_bf16(const float* __restrict__ in,
                                                __bf16* __restrict__ out, int n) {
    int i = blockIdx.x * 256 + threadIdx.x;
    if (i < n) out[i] = f2bf(in[i]);
}

// ---------------------------------------------------------------------------
// P1: x (B,256,N) fp32  ->  XT (B,N,256) bf16   (coalesced reads across lanes,
//     each lane emits its own K-row as 16B vector stores)
// ---------------------------------------------------------------------------
__global__ void __launch_bounds__(256) cvt_transpose_x(const float* __restrict__ x,
                                                       __bf16* __restrict__ xt) {
    int idx = blockIdx.x * 256 + threadIdx.x;           // b*N + n, total 8*3136
    if (idx >= 8 * NPIX) return;
    int b = idx / NPIX, n = idx % NPIX;
    const float* src = x + (size_t)b * 256 * NPIX + n;
    __bf16* dst = xt + (size_t)idx * 256;
    #pragma unroll 4
    for (int k = 0; k < 256; k += 8) {
        v8bf v;
        #pragma unroll
        for (int e = 0; e < 8; ++e) v[e] = f2bf(src[(size_t)(k + e) * NPIX]);
        *(v8bf*)(dst + k) = v;
    }
}

// ---------------------------------------------------------------------------
// WMMA fragment helpers (K-contiguous order used by this file for A and B, wave32)
//   A 16x32 bf16: lane<16 = row M=lane, elems = K[half*16..+16)  (K-contiguous)
//   B 32x16 bf16: lane<16 = col N=lane, elems = K[half*16..+16)  (K-contiguous)
// ---------------------------------------------------------------------------
__device__ __forceinline__ v16bf load_afrag(const __bf16* __restrict__ row, int half) {
    v8bf lo = *(const v8bf*)(row + half * 16);
    v8bf hi = *(const v8bf*)(row + half * 16 + 8);
    return __builtin_shufflevector(lo, hi, 0, 1, 2, 3, 4, 5, 6, 7,
                                           8, 9, 10, 11, 12, 13, 14, 15);
}

// ---------------------------------------------------------------------------
// K1: qkv GEMM  out[b,m,n] = sum_k W[m,k] * X[b,k,n]
//     M=768, K=256, N=3136.  4M x 2N register blocking per wave.
//     A = bf16 weights (M,K) row-major; B = XT bf16 (N,K) row-major.
// ---------------------------------------------------------------------------
__global__ void __launch_bounds__(128) qkv_gemm(const __bf16* __restrict__ xt,
                                                const __bf16* __restrict__ wbf,
                                                float* __restrict__ out) {
    const int N = NPIX, K = 256, M = 768;
    const int wave = threadIdx.x >> 5;
    const int lane = threadIdx.x & 31;
    const int half = lane >> 4;
    const int l15  = lane & 15;

    const int b      = blockIdx.x / 3;
    const int m0     = ((blockIdx.x % 3) * 4 + wave) * 64;  // 4 mtiles of 16
    const int ntile0 = blockIdx.y * 32;                     // 2 ntiles of 16

    const __bf16* __restrict__ XTb = xt + (size_t)b * N * K;

    v8f acc[4][2];
    #pragma unroll
    for (int i = 0; i < 4; ++i)
        #pragma unroll
        for (int j = 0; j < 2; ++j) acc[i][j] = (v8f){};

    const __bf16* brow0 = XTb + (size_t)(ntile0 + l15) * K + half * 16;
    const __bf16* brow1 = XTb + (size_t)(ntile0 + 16 + l15) * K + half * 16;
    const __bf16* arow  = wbf + (size_t)(m0 + l15) * K;

    for (int k0 = 0; k0 < K; k0 += 32) {
        v16bf bf0 = *(const v16bf*)(brow0 + k0);
        v16bf bf1 = *(const v16bf*)(brow1 + k0);
        #pragma unroll
        for (int i = 0; i < 4; ++i) {
            v16bf a = load_afrag(arow + (size_t)i * 16 * K + k0, half);
            acc[i][0] = __builtin_amdgcn_wmma_f32_16x16x32_bf16(false, a, false, bf0,
                                                                (short)0, acc[i][0], false, false);
            acc[i][1] = __builtin_amdgcn_wmma_f32_16x16x32_bf16(false, a, false, bf1,
                                                                (short)0, acc[i][1], false, false);
        }
    }
    // C/D layout: VGPR r -> row +8*half + r, col = l15 (+16 for j=1)
    float* obase = out + (size_t)b * M * N;
    #pragma unroll
    for (int i = 0; i < 4; ++i)
        #pragma unroll
        for (int j = 0; j < 2; ++j) {
            float* orow = obase + (size_t)(m0 + i * 16 + 8 * half) * N + ntile0 + j * 16 + l15;
            #pragma unroll
            for (int r = 0; r < 8; ++r) orow[(size_t)r * N] = acc[i][j][r];
        }
}

// ---------------------------------------------------------------------------
// K2: depthwise 5x5, pad 2, 768 channels
// ---------------------------------------------------------------------------
__global__ void __launch_bounds__(256) dwconv(const float* __restrict__ in,
                                              const float* __restrict__ w,
                                              float* __restrict__ out) {
    const size_t total = (size_t)8 * 768 * NPIX;
    size_t idx = (size_t)blockIdx.x * 256 + threadIdx.x;
    if (idx >= total) return;
    int xw = (int)(idx % WW); size_t t = idx / WW;
    int y  = (int)(t % HH);   t /= HH;
    int c  = (int)(t % 768);
    int b  = (int)(t / 768);
    const float* src = in + ((size_t)b * 768 + c) * NPIX;
    const float* wf  = w + c * 25;
    float s = 0.f;
    #pragma unroll
    for (int dy = 0; dy < 5; ++dy) {
        int yy = y + dy - 2;
        if ((unsigned)yy < (unsigned)HH) {
            #pragma unroll
            for (int dx = 0; dx < 5; ++dx) {
                int xx = xw + dx - 2;
                if ((unsigned)xx < (unsigned)WW)
                    s += wf[dy * 5 + dx] * src[yy * WW + xx];
            }
        }
    }
    out[idx] = s;
}

// ---------------------------------------------------------------------------
// K3: grouped 1x1, 96 groups of 8->8, in place on dw buffer
// ---------------------------------------------------------------------------
__global__ void __launch_bounds__(256) pwconv(float* __restrict__ buf,
                                              const float* __restrict__ w) {
    const size_t total = (size_t)8 * 96 * NPIX;
    size_t idx = (size_t)blockIdx.x * 256 + threadIdx.x;
    if (idx >= total) return;
    int n = (int)(idx % NPIX); size_t t = idx / NPIX;
    int g = (int)(t % 96);
    int b = (int)(t / 96);
    float* base = buf + ((size_t)b * 768 + (size_t)g * 8) * NPIX + n;
    float in8[8], o8[8];
    #pragma unroll
    for (int i = 0; i < 8; ++i) in8[i] = base[(size_t)i * NPIX];
    const float* wg = w + g * 64;
    #pragma unroll
    for (int o = 0; o < 8; ++o) {
        float s = 0.f;
        #pragma unroll
        for (int i = 0; i < 8; ++i) s += wg[o * 8 + i] * in8[i];
        o8[o] = s;
    }
    #pragma unroll
    for (int o = 0; o < 8; ++o) base[(size_t)o * NPIX] = o8[o];
}

// ---------------------------------------------------------------------------
// K4a: kv[b,h] = sum_n k9(n) (outer) v9(n)   (9x9 per (b,h))
// ---------------------------------------------------------------------------
__global__ void __launch_bounds__(256) kv_reduce(const float* __restrict__ qkv,
                                                 const float* __restrict__ dwb,
                                                 const float* __restrict__ pos,
                                                 const float* __restrict__ scale1,
                                                 float* __restrict__ kvout) {
    const int N = NPIX;
    int bh = blockIdx.x;
    int b = bh >> 6, h = bh & 63;
    const float* src = (h < 32)
        ? qkv + ((size_t)b * 768 + (size_t)h * 24) * N
        : dwb + ((size_t)b * 768 + (size_t)(h - 32) * 24) * N;
    const float* kb = src + (size_t)8 * N;
    const float* vb = src + (size_t)16 * N;
    const float* pb = pos + (size_t)(h * 8) * N;
    const float s1 = scale1[0];

    float acc[81];
    #pragma unroll
    for (int i = 0; i < 81; ++i) acc[i] = 0.f;

    for (int n = threadIdx.x; n < N; n += 256) {
        float kk[8], k9[9], v9[9];
        float ss = 0.f;
        #pragma unroll
        for (int d = 0; d < 8; ++d) {
            kk[d] = kb[(size_t)d * N + n] + pb[(size_t)d * N + n];
            ss += kk[d] * kk[d];
        }
        float r1 = 1.f / (sqrtf(ss) + EPSF);
        ss = 0.f;
        #pragma unroll
        for (int d = 0; d < 8; ++d) {
            float u = kk[d] * r1; u *= u;
            kk[d] = u; ss += u * u;
        }
        float r2 = 1.f / (sqrtf(ss) + EPSF);
        #pragma unroll
        for (int d = 0; d < 8; ++d) k9[d] = kk[d] * r2;
        k9[8] = s1;
        #pragma unroll
        for (int d = 0; d < 8; ++d) v9[d] = vb[(size_t)d * N + n];
        v9[8] = 1.f;
        #pragma unroll
        for (int i = 0; i < 9; ++i)
            #pragma unroll
            for (int j = 0; j < 9; ++j)
                acc[i * 9 + j] = fmaf(k9[i], v9[j], acc[i * 9 + j]);
    }

    __shared__ float red[8][81];
    #pragma unroll 1
    for (int t = 0; t < 81; ++t) {
        float v = acc[t];
        for (int off = 16; off > 0; off >>= 1) v += __shfl_xor(v, off, 32);
        if ((threadIdx.x & 31) == 0) red[threadIdx.x >> 5][t] = v;
    }
    __syncthreads();
    if (threadIdx.x < 81) {
        float s = 0.f;
        #pragma unroll
        for (int wv = 0; wv < 8; ++wv) s += red[wv][threadIdx.x];
        kvout[(size_t)bh * 81 + threadIdx.x] = s;
    }
}

// ---------------------------------------------------------------------------
// K4b: out9 = q9 . kv ; divide; + gelu(bn(v));
//      write bf16 transposed (B, N, 512) -> single 16B store per thread
// ---------------------------------------------------------------------------
__global__ void __launch_bounds__(256) attn_out(const float* __restrict__ qkv,
                                                const float* __restrict__ dwb,
                                                const float* __restrict__ kv,
                                                const float* __restrict__ scale1,
                                                const float* __restrict__ bg,
                                                const float* __restrict__ bb,
                                                const float* __restrict__ bm,
                                                const float* __restrict__ bv,
                                                __bf16* __restrict__ attT) {
    const int N = NPIX;
    int bh = blockIdx.y;
    int b = bh >> 6, h = bh & 63;
    int n = blockIdx.x * 256 + threadIdx.x;
    if (n >= N) return;
    const float* src = (h < 32)
        ? qkv + ((size_t)b * 768 + (size_t)h * 24) * N
        : dwb + ((size_t)b * 768 + (size_t)(h - 32) * 24) * N;

    float q[8], q9[9], ss = 0.f;
    #pragma unroll
    for (int d = 0; d < 8; ++d) { q[d] = src[(size_t)d * N + n]; ss += q[d] * q[d]; }
    float r1 = 1.f / (sqrtf(ss) + EPSF);
    ss = 0.f;
    #pragma unroll
    for (int d = 0; d < 8; ++d) { float u = q[d] * r1; u *= u; q[d] = u; ss += u * u; }
    float r2 = 1.f / (sqrtf(ss) + EPSF);
    #pragma unroll
    for (int d = 0; d < 8; ++d) q9[d] = q[d] * r2;
    q9[8] = scale1[0];

    const float* kvp = kv + (size_t)bh * 81;
    float o[9];
    #pragma unroll
    for (int d = 0; d < 9; ++d) {
        float s = 0.f;
        #pragma unroll
        for (int i = 0; i < 9; ++i) s = fmaf(q9[i], kvp[i * 9 + d], s);
        o[d] = s;
    }
    float inv = 1.f / (o[8] + EPSF);

    v8bf res;
    #pragma unroll
    for (int d = 0; d < 8; ++d) {
        float vv = src[(size_t)(16 + d) * N + n];
        float sc = bg[d] * rsqrtf(bv[d] + 1e-5f);
        float fm = (vv - bm[d]) * sc + bb[d];
        fm = 0.5f * fm * (1.f + erff(fm * 0.70710678118654752440f));  // exact gelu
        res[d] = f2bf(o[d] * inv + fm);
    }
    // proj-B layout: attT[b][n][h*8 + d], contiguous 8 bf16 = one 16B store
    *(v8bf*)(attT + ((size_t)b * N + n) * 512 + (size_t)h * 8) = res;
}

// ---------------------------------------------------------------------------
// K5: proj GEMM 256x512 x (512 x N) per batch + fused BN -> d_out
//     A = bf16 w_proj (256,512); B = attT bf16 (N,512) row-major
// ---------------------------------------------------------------------------
__global__ void __launch_bounds__(128) proj_gemm(const __bf16* __restrict__ attT,
                                                 const __bf16* __restrict__ wbf,
                                                 const float* __restrict__ gamma,
                                                 const float* __restrict__ beta,
                                                 const float* __restrict__ mean,
                                                 const float* __restrict__ var,
                                                 float* __restrict__ out) {
    const int N = NPIX, K = 512, M = 256;
    const int wave = threadIdx.x >> 5;
    const int lane = threadIdx.x & 31;
    const int half = lane >> 4;
    const int l15  = lane & 15;

    const int b      = blockIdx.x;           // 0..7
    const int m0     = wave * 64;            // 4 waves cover M=256
    const int ntile0 = blockIdx.y * 32;      // 2 ntiles of 16

    const __bf16* __restrict__ Bb = attT + (size_t)b * N * K;

    v8f acc[4][2];
    #pragma unroll
    for (int i = 0; i < 4; ++i)
        #pragma unroll
        for (int j = 0; j < 2; ++j) acc[i][j] = (v8f){};

    const __bf16* brow0 = Bb + (size_t)(ntile0 + l15) * K + half * 16;
    const __bf16* brow1 = Bb + (size_t)(ntile0 + 16 + l15) * K + half * 16;
    const __bf16* arow  = wbf + (size_t)(m0 + l15) * K;

    for (int k0 = 0; k0 < K; k0 += 32) {
        v16bf bf0 = *(const v16bf*)(brow0 + k0);
        v16bf bf1 = *(const v16bf*)(brow1 + k0);
        #pragma unroll
        for (int i = 0; i < 4; ++i) {
            v16bf a = load_afrag(arow + (size_t)i * 16 * K + k0, half);
            acc[i][0] = __builtin_amdgcn_wmma_f32_16x16x32_bf16(false, a, false, bf0,
                                                                (short)0, acc[i][0], false, false);
            acc[i][1] = __builtin_amdgcn_wmma_f32_16x16x32_bf16(false, a, false, bf1,
                                                                (short)0, acc[i][1], false, false);
        }
    }
    float* obase = out + (size_t)b * M * N;
    #pragma unroll
    for (int i = 0; i < 4; ++i) {
        #pragma unroll
        for (int r = 0; r < 8; ++r) {
            int row = m0 + i * 16 + 8 * half + r;
            float s = gamma[row] * rsqrtf(var[row] + 1e-5f);
            float t = beta[row] - mean[row] * s;
            #pragma unroll
            for (int j = 0; j < 2; ++j)
                obase[(size_t)row * N + ntile0 + j * 16 + l15] = acc[i][j][r] * s + t;
        }
    }
}

// ---------------------------------------------------------------------------
extern "C" void kernel_launch(void* const* d_in, const int* in_sizes, int n_in,
                              void* d_out, int out_size, void* d_ws, size_t ws_size,
                              hipStream_t stream) {
    const float* x      = (const float*)d_in[0];
    const float* w_qkv  = (const float*)d_in[1];
    const float* w_dw   = (const float*)d_in[2];
    const float* w_pw   = (const float*)d_in[3];
    const float* pos    = (const float*)d_in[4];
    const float* s1     = (const float*)d_in[5];
    const float* bng    = (const float*)d_in[6];
    const float* bnb    = (const float*)d_in[7];
    const float* bnm    = (const float*)d_in[8];
    const float* bnv    = (const float*)d_in[9];
    const float* w_proj = (const float*)d_in[10];
    const float* pg     = (const float*)d_in[11];
    const float* pb     = (const float*)d_in[12];
    const float* pm     = (const float*)d_in[13];
    const float* pv     = (const float*)d_in[14];
    float* out = (float*)d_out;

    const size_t N = NPIX;
    float*  qkv   = (float*)d_ws;                           // 8*768*N f32
    float*  dwb   = qkv + (size_t)8 * 768 * N;              // 8*768*N f32
    float*  kvb   = dwb + (size_t)8 * 768 * N;              // 8*64*81 f32
    __bf16* xt    = (__bf16*)(kvb + (size_t)8 * 64 * 81);   // 8*N*256 bf16
    __bf16* attT  = xt + (size_t)8 * N * 256;               // 8*N*512 bf16
    __bf16* wqb   = attT + (size_t)8 * N * 512;             // 768*256 bf16
    __bf16* wpb   = wqb + (size_t)768 * 256;                // 256*512 bf16

    // P0: weight conversions
    cvt_bf16<<<dim3((768 * 256 + 255) / 256), dim3(256), 0, stream>>>(w_qkv, wqb, 768 * 256);
    cvt_bf16<<<dim3((256 * 512 + 255) / 256), dim3(256), 0, stream>>>(w_proj, wpb, 256 * 512);

    // P1: x -> XT bf16 (B,N,256)
    cvt_transpose_x<<<dim3((8 * NPIX + 255) / 256), dim3(256), 0, stream>>>(x, xt);

    // K1: qkv GEMM (WMMA bf16, 4Mx2N blocking)
    qkv_gemm<<<dim3(8 * 3, 98), dim3(128), 0, stream>>>(xt, wqb, qkv);

    // K2: depthwise 5x5
    {
        size_t total = (size_t)8 * 768 * NPIX;
        dwconv<<<dim3((unsigned)((total + 255) / 256)), dim3(256), 0, stream>>>(qkv, w_dw, dwb);
    }

    // K3: grouped pointwise (in place on dwb)
    {
        size_t total = (size_t)8 * 96 * NPIX;
        pwconv<<<dim3((unsigned)((total + 255) / 256)), dim3(256), 0, stream>>>(dwb, w_pw);
    }

    // K4a: kv 9x9 reductions
    kv_reduce<<<dim3(8 * 64), dim3(256), 0, stream>>>(qkv, dwb, pos, s1, kvb);

    // K4b: attention apply + BN/GELU residual -> attT bf16 (B,N,512)
    attn_out<<<dim3((NPIX + 255) / 256, 8 * 64), dim3(256), 0, stream>>>(
        qkv, dwb, kvb, s1, bng, bnb, bnm, bnv, attT);

    // K5: proj GEMM (WMMA bf16) + fused BN -> d_out
    proj_gemm<<<dim3(8, 98), dim3(128), 0, stream>>>(attT, wpb, pg, pb, pm, pv, out);
}
